// NumberTranslationModel_61916248539389
// MI455X (gfx1250) — matrix operand order repaired and checked by reading, not compiled, not run on hardware
//
#include <hip/hip_runtime.h>
#include <hip/hip_bf16.h>

typedef __attribute__((ext_vector_type(16))) _Float16     v16h;
typedef __attribute__((ext_vector_type(8)))  float        v8f;
typedef __attribute__((ext_vector_type(4)))  unsigned int u32x4;
typedef __attribute__((ext_vector_type(8)))  unsigned int u32x8;

#define S_IN   128
#define BSZ    512
#define S_OUTT 64
#define H_ENC  256
#define DEC_HD 1024
#define EMB    64
#define NVOC   128

// ---------------------------------------------------------------------------
// helpers
// ---------------------------------------------------------------------------
union FragCvt { u32x4 u[2]; v16h v; };

__device__ inline v16h ld_frag(const _Float16* p0, const _Float16* p1) {
  FragCvt f;
  f.u[0] = *reinterpret_cast<const u32x4*>(p0);
  f.u[1] = *reinterpret_cast<const u32x4*>(p1);
  return f.v;
}

// ---------------------------------------------------------------------------
// TDM 2-D tile load: rows x 32 f16 elements, row stride = K elements,
// DMA'd from global into LDS at ldsDst. Descriptor per CDNA5 ISA ch.8
// (D# groups 0..3); issued by one wave, tracked with TENSORcnt.
// The flat address of an LDS object is {SHARED_BASE, as3_offset}; hardware
// truncates LDS flat addresses to addr[31:0], so the low 32 bits of the flat
// pointer are the wave-relative LDS byte address required by D#.lds_addr.
// The pointer is also passed as an asm operand so the LDS buffer escapes and
// the "memory" clobber keeps the consuming ds_loads alive.
// ---------------------------------------------------------------------------
__device__ inline void tdm_load_tile(const _Float16* ldsDst, const _Float16* base,
                                     int row0, int K, int k0, int rows)
{
  unsigned ldsOff = (unsigned)(unsigned long long)(const void*)ldsDst;
  unsigned long long ga = (unsigned long long)(const void*)base +
      ((unsigned long long)(unsigned)row0 * (unsigned)K + (unsigned)k0) * 2ull;
  u32x4 g0 = {};
  g0[0] = 1u;                                            // count=1, user mode
  g0[1] = ldsOff;                                        // lds_addr (bytes)
  g0[2] = (unsigned)ga;                                  // global_addr[31:0]
  g0[3] = (unsigned)((ga >> 32) & 0x01FFFFFFu) | (2u << 30);  // addr[56:32] | type=2
  u32x8 g1 = {};
  g1[0] = (1u << 16);                  // workgroup_mask=0, data_size=1 (2 bytes)
  g1[1] = 0xFFFF0000u;                 // tensor_dim0 = 0x7FFFFFFF (lo16 here)
  g1[2] = 0xFFFF7FFFu;                 // tensor_dim0 hi16 | tensor_dim1 lo16
  g1[3] = 0x00007FFFu | (32u << 16);   // tensor_dim1 hi16 | tile_dim0 = 32
  g1[4] = (unsigned)rows;              // tile_dim1 = rows, tile_dim2 = 0 (2-D)
  g1[5] = (unsigned)K;                 // tensor_dim0_stride[31:0] (element units)
  g1[6] = 0u;                          // stride hi | tensor_dim1_stride lo (unused)
  g1[7] = 0u;
  u32x4 g2 = {};
  u32x4 g3 = {};
  asm volatile("tensor_load_to_lds %0, %1, %2, %3"
               :: "s"(g0), "s"(g1), "s"(g2), "s"(g3), "v"(ldsDst)
               : "memory");
}

struct Smem {
  _Float16 A[2][128 * 32];   // ping/pong A tiles (8 KB each)
  _Float16 B[2][64 * 32];    // ping/pong B tiles (4 KB each)
};

// ---------------------------------------------------------------------------
// Fused gate GEMM:  G[M=512, N] = Xa[512,Kx] @ Wx[N,Kx]^T + Ha[512,Kh] @ Wh[N,Kh]^T + bias
// block = 256 threads (8 wave32), tile 128x64, each wave 16x64 (4 wmma accums).
// K staged in 32-wide chunks by the Tensor Data Mover, double-buffered so the
// DMA of chunk i+1 overlaps the WMMA work of chunk i. Kx may be 0 (projection).
// ---------------------------------------------------------------------------
__global__ __launch_bounds__(256)
void gate_gemm(const _Float16* __restrict__ Xa, int Kx,
               const _Float16* __restrict__ Ha, int Kh,
               const _Float16* __restrict__ Wx,
               const _Float16* __restrict__ Wh,
               const float* __restrict__ bias,
               float* __restrict__ G, int N)
{
  __shared__ Smem smem;

  const int tid  = threadIdx.x;
  const int wave = tid >> 5;
  const int lane = tid & 31;
  const int rowBase = blockIdx.y * 128;
  const int nBase   = blockIdx.x * 64;
  const int cX  = Kx >> 5;
  const int nch = cX + (Kh >> 5);

  v8f acc[4] = {};

  auto issue = [&](int ci) {
    const int buf = ci & 1;
    const _Float16* A;
    const _Float16* W;
    int K, k0;
    if (ci < cX) { A = Xa; W = Wx; K = Kx; k0 = ci << 5; }
    else         { A = Ha; W = Wh; K = Kh; k0 = (ci - cX) << 5; }
    tdm_load_tile(&smem.A[buf][0], A, rowBase, K, k0, 128);
    tdm_load_tile(&smem.B[buf][0], W, nBase,   K, k0, 64);
  };

  if (wave == 0) issue(0);

  for (int ci = 0; ci < nch; ++ci) {
    if (wave == 0) {
      if (ci + 1 < nch) {
        issue(ci + 1);                                // overlap DMA with compute
        __builtin_amdgcn_s_wait_tensorcnt((short)2);  // chunk ci complete
      } else {
        __builtin_amdgcn_s_wait_tensorcnt((short)0);
      }
    }
    __syncthreads();   // publish TDM-written LDS to all 8 waves

    const int buf  = ci & 1;
    // A fragment (ISA 16-bit A 16x32 layout): lane&15 = row,
    // lanes 0-15 hold K 0-7 & 16-23, lanes 16-31 hold K 8-15 & 24-31
    const int rowA = (wave << 4) + (lane & 15);
    const int kb2  = (lane < 16) ? 0 : 8;
    const int kbB  = (lane < 16) ? 0 : 8;
    const _Float16* ap = &smem.A[buf][rowA * 32];
    v16h a = ld_frag(ap + kb2, ap + kb2 + 16);
#pragma unroll
    for (int ns = 0; ns < 4; ++ns) {
      // B fragment: lane&15 = column n, same K split as A (0-7 & 16-23 / 8-15 & 24-31)
      const _Float16* bp = &smem.B[buf][((ns << 4) + (lane & 15)) * 32 + kbB];
      v16h b = ld_frag(bp, bp + 16);
      acc[ns] = __builtin_amdgcn_wmma_f32_16x16x32_f16(
          false, a, false, b, (short)0, acc[ns], false, false);
    }
    __syncthreads();   // tile consumed; buffer may be overwritten next iter
  }

  // C/D layout: VGPR i -> M = i + (lane>=16 ? 8 : 0); N = lane&15
  const int mhi = (lane >= 16) ? 8 : 0;
#pragma unroll
  for (int ns = 0; ns < 4; ++ns) {
    const int col = nBase + (ns << 4) + (lane & 15);
    const float bv = bias[col];
#pragma unroll
    for (int i = 0; i < 8; ++i) {
      const int row = rowBase + (wave << 4) + i + mhi;
      G[(size_t)row * N + col] = acc[ns][i] + bv;
    }
  }
}

// ---------------------------------------------------------------------------
// LSTM pointwise:  i,f,g,o gate chunks -> update c, h; emit f16 copies
// ---------------------------------------------------------------------------
__global__ void lstm_point(const float* __restrict__ G, int Hd, int n,
                           float* __restrict__ h, float* __restrict__ c,
                           _Float16* __restrict__ h16,
                           _Float16* __restrict__ seq_out, int seq_stride,
                           _Float16* __restrict__ c16)
{
  int idx = blockIdx.x * blockDim.x + threadIdx.x;
  if (idx >= n) return;
  int b = idx / Hd;
  int j = idx - b * Hd;
  const float* g = G + (size_t)b * 4 * Hd;
  float gi = g[j], gf = g[Hd + j], gg = g[2 * Hd + j], go = g[3 * Hd + j];
  float si = 1.f / (1.f + __expf(-gi));
  float sf = 1.f / (1.f + __expf(-gf));
  float so = 1.f / (1.f + __expf(-go));
  float cn = sf * c[idx] + si * tanhf(gg);
  float hn = so * tanhf(cn);
  c[idx] = cn;
  h[idx] = hn;
  h16[idx] = (_Float16)hn;
  if (seq_out) seq_out[(size_t)b * seq_stride + j] = (_Float16)hn;
  if (c16)     c16[idx] = (_Float16)cn;
}

__global__ void scatter_state(const float* __restrict__ src, float* __restrict__ dst,
                              int Hd, int colOff, int n)
{
  int idx = blockIdx.x * blockDim.x + threadIdx.x;
  if (idx >= n) return;
  int b = idx / Hd, j = idx - b * Hd;
  dst[(size_t)b * (DEC_HD) + colOff + j] = src[idx];
}

__global__ void f32_to_f16(const float* __restrict__ s, _Float16* __restrict__ d, int n)
{
  int i = blockIdx.x * blockDim.x + threadIdx.x;
  if (i < n) d[i] = (_Float16)s[i];
}

__global__ void embed_in(const int* __restrict__ toks, const float* __restrict__ emb,
                         _Float16* __restrict__ out, int n)  // n = S_IN*BSZ*EMB
{
  int i = blockIdx.x * blockDim.x + threadIdx.x;
  if (i >= n) return;
  int e = i & (EMB - 1);
  int r = i >> 6;
  out[i] = (_Float16)emb[toks[r] * EMB + e];
}

__global__ void embed_dec(const int* __restrict__ out_toks, const float* __restrict__ emb,
                          _Float16* __restrict__ eall)  // [63, B, EMB], token = out_toks[b, t]
{
  int i = blockIdx.x * blockDim.x + threadIdx.x;
  if (i >= (S_OUTT - 1) * BSZ * EMB) return;
  int e  = i & (EMB - 1);
  int tb = i >> 6;
  int b  = tb & (BSZ - 1);
  int t  = tb >> 9;
  eall[i] = (_Float16)emb[out_toks[b * S_OUTT + t] * EMB + e];
}

// ---------------------------------------------------------------------------
// argmax + log-softmax target gather; one wave32 per row of 128 logits
// ---------------------------------------------------------------------------
__global__ __launch_bounds__(256)
void argmax_lp(const float* __restrict__ logits, const int* __restrict__ out_toks,
               int t, float* __restrict__ dec_out, float* __restrict__ cum)
{
  const int lane = threadIdx.x & 31;
  const int b    = blockIdx.x * 8 + (threadIdx.x >> 5);
  const float* row = logits + (size_t)b * NVOC;

  float vals[4];
  float vmax = -3.402823466e38f;
  int   imax = 0;
#pragma unroll
  for (int q = 0; q < 4; ++q) {
    int col = lane + (q << 5);
    float v = row[col];
    vals[q] = v;
    if (v > vmax) { vmax = v; imax = col; }   // ascending col: strict > keeps first
  }
#pragma unroll
  for (int off = 16; off >= 1; off >>= 1) {
    float ov = __shfl_xor(vmax, off, 32);
    int   oi = __shfl_xor(imax, off, 32);
    if (ov > vmax || (ov == vmax && oi < imax)) { vmax = ov; imax = oi; }
  }
  float s = 0.f;
#pragma unroll
  for (int q = 0; q < 4; ++q) s += __expf(vals[q] - vmax);
#pragma unroll
  for (int off = 16; off >= 1; off >>= 1) s += __shfl_xor(s, off, 32);

  if (lane == 0) {
    int tn = out_toks[b * S_OUTT + t + 1];
    dec_out[t * BSZ + b] = (float)imax;
    cum[b] += row[tn] - vmax - logf(s);
  }
}

// ---------------------------------------------------------------------------
// host orchestration (graph-capture safe: only async ops on `stream`)
// ---------------------------------------------------------------------------
extern "C" void kernel_launch(void* const* d_in, const int* in_sizes, int n_in,
                              void* d_out, int out_size, void* d_ws, size_t ws_size,
                              hipStream_t stream)
{
  (void)in_sizes; (void)n_in; (void)out_size; (void)ws_size;

  const int*   in_toks  = (const int*)d_in[0];
  const int*   out_toks = (const int*)d_in[1];
  const float* in_emb   = (const float*)d_in[2];
  const float* eWih0    = (const float*)d_in[3];
  const float* eWhh0    = (const float*)d_in[4];
  const float* eb0      = (const float*)d_in[5];
  const float* eWih1    = (const float*)d_in[6];
  const float* eWhh1    = (const float*)d_in[7];
  const float* eb1      = (const float*)d_in[8];
  const float* out_emb  = (const float*)d_in[9];
  const float* dWih     = (const float*)d_in[10];
  const float* dWhh     = (const float*)d_in[11];
  const float* db       = (const float*)d_in[12];
  const float* pW       = (const float*)d_in[13];
  const float* pb       = (const float*)d_in[14];

  char* ws = (char*)d_ws;
  size_t off = 0;
  auto alloc = [&](size_t bytes) -> char* {
    char* p = ws + off;
    off += (bytes + 255) & ~(size_t)255;
    return p;
  };

  // f16 weights
  _Float16* w_ih0 = (_Float16*)alloc((size_t)2 * 1024 * 64 * 2);
  _Float16* w_hh0 = (_Float16*)alloc((size_t)2 * 1024 * 256 * 2);
  _Float16* w_ih1 = (_Float16*)alloc((size_t)2 * 1024 * 512 * 2);
  _Float16* w_hh1 = (_Float16*)alloc((size_t)2 * 1024 * 256 * 2);
  _Float16* dw_ih = (_Float16*)alloc((size_t)4096 * 64 * 2);
  _Float16* dw_hh = (_Float16*)alloc((size_t)4096 * 1024 * 2);
  _Float16* projW = (_Float16*)alloc((size_t)128 * 1024 * 2);
  // activations
  _Float16* x0    = (_Float16*)alloc((size_t)S_IN * BSZ * EMB * 2);
  _Float16* seq   = (_Float16*)alloc((size_t)S_IN * BSZ * 512 * 2);   // layer-0 bi output
  _Float16* eall  = (_Float16*)alloc((size_t)(S_OUTT - 1) * BSZ * EMB * 2);
  float*    gates = (float*)alloc((size_t)BSZ * 4096 * 4);
  float*    h0    = (float*)alloc((size_t)BSZ * DEC_HD * 4);
  float*    c0    = (float*)alloc((size_t)BSZ * DEC_HD * 4);
  float*    h32   = (float*)alloc((size_t)BSZ * H_ENC * 4);
  float*    c32   = (float*)alloc((size_t)BSZ * H_ENC * 4);
  _Float16* h16   = (_Float16*)alloc((size_t)BSZ * DEC_HD * 2);
  _Float16* c16   = (_Float16*)alloc((size_t)BSZ * DEC_HD * 2);
  float*    logits = (float*)alloc((size_t)BSZ * NVOC * 4);

  auto cvt = [&](const float* s, _Float16* d, int n) {
    f32_to_f16<<<(n + 255) / 256, 256, 0, stream>>>(s, d, n);
  };
  cvt(eWih0, w_ih0, 2 * 1024 * 64);
  cvt(eWhh0, w_hh0, 2 * 1024 * 256);
  cvt(eWih1, w_ih1, 2 * 1024 * 512);
  cvt(eWhh1, w_hh1, 2 * 1024 * 256);
  cvt(dWih,  dw_ih, 4096 * 64);
  cvt(dWhh,  dw_hh, 4096 * 1024);
  cvt(pW,    projW, 128 * 1024);

  embed_in<<<(S_IN * BSZ * EMB + 255) / 256, 256, 0, stream>>>(in_toks, in_emb, x0,
                                                               S_IN * BSZ * EMB);
  embed_dec<<<((S_OUTT - 1) * BSZ * EMB + 255) / 256, 256, 0, stream>>>(out_toks, out_emb, eall);

  // ---- encoder: 2-layer bidirectional LSTM ----
  const int ptN = (BSZ * H_ENC + 255) / 256;
  for (int layer = 0; layer < 2; ++layer) {
    const _Float16* Wx = layer ? w_ih1 : w_ih0;
    const _Float16* Wh = layer ? w_hh1 : w_hh0;
    const float* bias_all = layer ? eb1 : eb0;
    const int Kx = layer ? 512 : 64;
    const _Float16* xin = layer ? seq : x0;
    for (int dir = 0; dir < 2; ++dir) {
      hipMemsetAsync(h32, 0, (size_t)BSZ * H_ENC * 4, stream);
      hipMemsetAsync(c32, 0, (size_t)BSZ * H_ENC * 4, stream);
      hipMemsetAsync(h16, 0, (size_t)BSZ * H_ENC * 2, stream);
      const _Float16* Wxd = Wx + (size_t)dir * 1024 * Kx;
      const _Float16* Whd = Wh + (size_t)dir * 1024 * H_ENC;
      const float*    bd  = bias_all + dir * 1024;
      for (int s = 0; s < S_IN; ++s) {
        const int step = dir ? (S_IN - 1 - s) : s;
        const _Float16* xt = xin + (size_t)step * BSZ * Kx;
        gate_gemm<<<dim3(16, 4), 256, 0, stream>>>(xt, Kx, h16, H_ENC, Wxd, Whd, bd,
                                                   gates, 1024);
        _Float16* sq = (layer == 0) ? (seq + (size_t)step * BSZ * 512 + dir * H_ENC)
                                    : nullptr;
        lstm_point<<<ptN, 256, 0, stream>>>(gates, H_ENC, BSZ * H_ENC,
                                            h32, c32, h16, sq, 512, nullptr);
      }
      const int colOff = (layer * 2 + dir) * H_ENC;  // PyTorch h_n ordering
      scatter_state<<<ptN, 256, 0, stream>>>(h32, h0, H_ENC, colOff, BSZ * H_ENC);
      scatter_state<<<ptN, 256, 0, stream>>>(c32, c0, H_ENC, colOff, BSZ * H_ENC);
    }
  }

  // ---- decoder: teacher-forced LSTMCell + projection on CELL state ----
  f32_to_f16<<<(BSZ * DEC_HD + 255) / 256, 256, 0, stream>>>(h0, h16, BSZ * DEC_HD);
  float* dout = (float*)d_out;
  float* cum  = dout + (size_t)(S_OUTT - 1) * BSZ;
  hipMemsetAsync(cum, 0, (size_t)BSZ * 4, stream);

  const int ptD = (BSZ * DEC_HD + 255) / 256;
  for (int t = 0; t < S_OUTT - 1; ++t) {
    gate_gemm<<<dim3(64, 4), 256, 0, stream>>>(eall + (size_t)t * BSZ * EMB, EMB,
                                               h16, DEC_HD, dw_ih, dw_hh, db,
                                               gates, 4096);
    lstm_point<<<ptD, 256, 0, stream>>>(gates, DEC_HD, BSZ * DEC_HD,
                                        h0, c0, h16, nullptr, 0, c16);
    gate_gemm<<<dim3(2, 4), 256, 0, stream>>>(nullptr, 0, c16, DEC_HD,
                                              nullptr, projW, pb, logits, NVOC);
    argmax_lp<<<64, 256, 0, stream>>>(logits, out_toks, t, dout, cum);
  }
}
